// GATActor_3788161155325
// MI455X (gfx1250) — matrix operand order repaired and checked by reading, not compiled, not run on hardware
//
#include <hip/hip_runtime.h>

typedef __attribute__((ext_vector_type(16))) __bf16 v16bf;
typedef __attribute__((ext_vector_type(8)))  float  v8f;

#define HEADS 4
#define HID   64
#define FEAT  256   // HEADS*HID
#define INDIM 128
#define OUTD  5
#define NEG_SLOPE 0.2f

static inline int cdiv(int a, int b) { return (a + b - 1) / b; }

// ---------- helpers ----------
__device__ __forceinline__ unsigned fkey(float f) {
    unsigned u = __float_as_uint(f);
    return (u & 0x80000000u) ? ~u : (u | 0x80000000u);
}
__device__ __forceinline__ float fdecode(unsigned k) {
    unsigned u = (k & 0x80000000u) ? (k & 0x7FFFFFFFu) : ~k;
    return __uint_as_float(u);
}
__device__ __forceinline__ float lrelu(float e) { return e > 0.f ? e : NEG_SLOPE * e; }

// LDS byte offset of a generic pointer to a __shared__ object
__device__ __forceinline__ unsigned lds_offset(const void* p) {
    return (unsigned)(unsigned long long)(__attribute__((address_space(3))) const void*)p;
}
// CDNA5 async global->LDS 128-bit copy (per-lane, ASYNCcnt-tracked)
__device__ __forceinline__ void async_copy_b128(unsigned ldsAddr, const void* gaddr) {
    asm volatile("global_load_async_to_lds_b128 %0, %1, off"
                 :: "v"(ldsAddr), "v"(gaddr) : "memory");
}
__device__ __forceinline__ void wait_async4() {   // all but the newest 4 copies landed
    asm volatile("s_wait_asynccnt 0x4" ::: "memory");
}

// ---------- f32 -> bf16 conversion ----------
__global__ void k_f32_to_bf16(const float* __restrict__ in, __bf16* __restrict__ out, long long n) {
    long long t = (long long)blockIdx.x * blockDim.x + threadIdx.x;
    if (t < n) out[t] = (__bf16)in[t];
}

// ---------- f32 [K x 256] -> bf16 transposed [256 x K] ----------
__global__ void k_transpose_bf16(const float* __restrict__ W, __bf16* __restrict__ Wt, int K) {
    int t = blockIdx.x * blockDim.x + threadIdx.x;
    if (t >= K * FEAT) return;
    int n = t / K, k = t % K;                 // consecutive t -> consecutive k: coalesced writes
    Wt[(size_t)n * K + k] = (__bf16)W[(size_t)k * FEAT + n];
}

// ---------- bf16 WMMA GEMM: C[nrows x 256] = A[nrows x K] * W[K x 256], W given transposed ----------
#define TR 128
#define TC 128
#define TK 32
__global__ __launch_bounds__(256) void k_gemm_wmma(
    const __bf16* __restrict__ A, const __bf16* __restrict__ Bt,  // Bt: [256][K]
    float* __restrict__ C, int nrows, int K)
{
    // ping/pong tiles; 80B row stride keeps 16-lane column access conflict-free
    __shared__ __align__(16) __bf16 As[2][TR][TK + 8];  // [buf][row][k]
    __shared__ __align__(16) __bf16 Bs[2][TC][TK + 8];  // [buf][col][k]

    const int tid  = threadIdx.x;
    const int wave = tid >> 5;
    const int lane = tid & 31;
    const int half = lane >> 4;
    const int ml   = lane & 15;

    const int rowBase = blockIdx.x * TR;
    const int colBase = blockIdx.y * TC;

    v8f acc[8];
#pragma unroll
    for (int i = 0; i < 8; ++i) acc[i] = v8f{0.f,0.f,0.f,0.f,0.f,0.f,0.f,0.f};

    // staging map: each thread moves 16 bf16 (two b128) of one A row and one B col.
    // OOB rows are clamped (their C rows are never stored) so every thread always
    // issues exactly 4 async copies per tile -> exact ASYNCcnt bookkeeping.
    const int sr = tid >> 1;             // tile row (A) / tile col (B)
    const int sc = (tid & 1) * 16;       // k offset within tile
    int srow = rowBase + sr; if (srow > nrows - 1) srow = nrows - 1;
    const __bf16* gA = A  + (size_t)srow * K + sc;
    const __bf16* gB = Bt + (size_t)(colBase + sr) * K + sc;
    const unsigned ldsA[2] = { lds_offset(&As[0][sr][sc]), lds_offset(&As[1][sr][sc]) };
    const unsigned ldsB[2] = { lds_offset(&Bs[0][sr][sc]), lds_offset(&Bs[1][sr][sc]) };

    auto issue_tile = [&](int kb, int b) {
        async_copy_b128(ldsA[b],      gA + kb);
        async_copy_b128(ldsA[b] + 16, gA + kb + 8);
        async_copy_b128(ldsB[b],      gB + kb);
        async_copy_b128(ldsB[b] + 16, gB + kb + 8);
    };
    const int am = wave * 16 + ml;
    auto compute_tile = [&](const __bf16 (*Asb)[TK + 8], const __bf16 (*Bsb)[TK + 8]) {
        // A fragment: lanes 0-15 K {0..7,16..23}, lanes 16-31 K {8..15,24..31} (two b128)
        v16bf a;
#pragma unroll
        for (int j = 0; j < 8; ++j) {
            int k0 = (j < 4 ? 2 * j : 16 + 2 * (j - 4)) + 8 * half;
            a[2 * j]     = Asb[am][k0];
            a[2 * j + 1] = Asb[am][k0 + 1];
        }
#pragma unroll
        for (int nb = 0; nb < 8; ++nb) {
            // B fragment: lanes 0-15 K {0..7,16..23}, lanes 16-31 K {8..15,24..31} (two b128)
            v16bf b;
            const __bf16* bp = &Bsb[nb * 16 + ml][8 * half];
#pragma unroll
            for (int i = 0; i < 16; ++i) b[i] = bp[i < 8 ? i : i + 8];
            acc[nb] = __builtin_amdgcn_wmma_f32_16x16x32_bf16(
                false, a, false, b, (short)0, acc[nb], false, false);
        }
    };

    // software pipeline, unrolled x2 (K/TK is even: 4 or 8 tiles)
    issue_tile(0, 0);
    for (int kb = 0; kb < K; kb += 2 * TK) {
        // phase 0: prefetch kb+TK into buf1, compute kb from buf0
        issue_tile(kb + TK, 1);
        wait_async4();
        __syncthreads();
        compute_tile(As[0], Bs[0]);
        __syncthreads();
        // phase 1: prefetch kb+2TK into buf0 (wrap on last; never consumed), compute kb+TK from buf1
        int nxt = (kb + 2 * TK < K) ? kb + 2 * TK : 0;
        issue_tile(nxt, 0);
        wait_async4();
        __syncthreads();
        compute_tile(As[1], Bs[1]);
        __syncthreads();
    }
    // trailing dummy prefetch drains at s_endpgm (implicit wait-idle)

    // C layout: VGPR r -> row r (lanes 0-15) / r+8 (lanes 16-31), col = lane%16
#pragma unroll
    for (int nb = 0; nb < 8; ++nb) {
        int col = colBase + nb * 16 + ml;
#pragma unroll
        for (int r = 0; r < 8; ++r) {
            int row = rowBase + wave * 16 + r + 8 * half;
            if (row < nrows) C[(size_t)row * FEAT + col] = acc[nb][r];
        }
    }
}

// ---------- per-node attention logits ----------
__global__ void k_alpha(const float* __restrict__ hlin,
                        const float* __restrict__ aS, const float* __restrict__ aD,
                        float* __restrict__ oS, float* __restrict__ oD, int n)
{
    int t = blockIdx.x * blockDim.x + threadIdx.x;
    if (t >= n * HEADS) return;
    int node = t >> 2, h = t & 3;
    const float* hp = hlin + (size_t)node * FEAT + h * HID;
    const float* s  = aS + h * HID;
    const float* d  = aD + h * HID;
    float accS = 0.f, accD = 0.f;
#pragma unroll 4
    for (int c = 0; c < HID; c += 4) {
        float4 hv = *reinterpret_cast<const float4*>(hp + c);
        float4 sv = *reinterpret_cast<const float4*>(s + c);
        float4 dv = *reinterpret_cast<const float4*>(d + c);
        accS += hv.x * sv.x + hv.y * sv.y + hv.z * sv.z + hv.w * sv.w;
        accD += hv.x * dv.x + hv.y * dv.y + hv.z * dv.z + hv.w * dv.w;
    }
    oS[t] = accS; oD[t] = accD;
}

// ---------- edge pass 1: segment max (order-preserving uint atomicMax) ----------
__global__ void k_edge_max(const int* __restrict__ src, const int* __restrict__ dst,
                           const float* __restrict__ as_, const float* __restrict__ ad_,
                           unsigned* __restrict__ smax, int e_raw, int n)
{
    int t = blockIdx.x * blockDim.x + threadIdx.x;
    int etot = e_raw + n;
    if (t >= etot * HEADS) return;
    int h = t & 3, ei = t >> 2;
    int s, d;
    if (ei < e_raw) { s = src[ei]; d = dst[ei]; } else { s = d = ei - e_raw; }
    float e = lrelu(as_[s * HEADS + h] + ad_[d * HEADS + h]);
    atomicMax(&smax[d * HEADS + h], fkey(e));
}

// ---------- edge pass 2: exp-sum ----------
__global__ void k_edge_sum(const int* __restrict__ src, const int* __restrict__ dst,
                           const float* __restrict__ as_, const float* __restrict__ ad_,
                           const unsigned* __restrict__ smax, float* __restrict__ ssum,
                           int e_raw, int n)
{
    int t = blockIdx.x * blockDim.x + threadIdx.x;
    int etot = e_raw + n;
    if (t >= etot * HEADS) return;
    int h = t & 3, ei = t >> 2;
    int s, d;
    if (ei < e_raw) { s = src[ei]; d = dst[ei]; } else { s = d = ei - e_raw; }
    float e = lrelu(as_[s * HEADS + h] + ad_[d * HEADS + h]);
    float m = fdecode(smax[d * HEADS + h]);
    atomicAdd(&ssum[d * HEADS + h], __expf(e - m));
}

// ---------- edge pass 3: weighted aggregation, one wave32 per edge ----------
__device__ __forceinline__ float edge_alpha(int s, int d, int h,
    const float* as_, const float* ad_, const unsigned* smax, const float* ssum)
{
    float e = lrelu(as_[s * HEADS + h] + ad_[d * HEADS + h]);
    float m = fdecode(smax[d * HEADS + h]);
    return __expf(e - m) / ssum[d * HEADS + h];
}

__global__ __launch_bounds__(256) void k_edge_agg(
    const int* __restrict__ src, const int* __restrict__ dst,
    const float* __restrict__ as_, const float* __restrict__ ad_,
    const unsigned* __restrict__ smax, const float* __restrict__ ssum,
    const float* __restrict__ hlin, float* __restrict__ agg, int e_raw, int n)
{
    int gwave = (int)((blockIdx.x * blockDim.x + threadIdx.x) >> 5);
    int lane  = threadIdx.x & 31;
    int etot  = e_raw + n;
    if (gwave >= etot) return;
    int s, d;
    if (gwave < e_raw) { s = src[gwave]; d = dst[gwave]; } else { s = d = gwave - e_raw; }

    int h1 = lane >> 4;      // channels 4*lane      -> heads 0/1
    int h2 = h1 + 2;         // channels 128+4*lane  -> heads 2/3
    float al1 = edge_alpha(s, d, h1, as_, ad_, smax, ssum);
    float al2 = edge_alpha(s, d, h2, as_, ad_, smax, ssum);

    int c1 = 4 * lane, c2 = 128 + 4 * lane;
    float4 v1 = *reinterpret_cast<const float4*>(hlin + (size_t)s * FEAT + c1);
    float4 v2 = *reinterpret_cast<const float4*>(hlin + (size_t)s * FEAT + c2);
    float* o1 = agg + (size_t)d * FEAT + c1;
    float* o2 = agg + (size_t)d * FEAT + c2;
    atomicAdd(o1 + 0, v1.x * al1); atomicAdd(o1 + 1, v1.y * al1);
    atomicAdd(o1 + 2, v1.z * al1); atomicAdd(o1 + 3, v1.w * al1);
    atomicAdd(o2 + 0, v2.x * al2); atomicAdd(o2 + 1, v2.y * al2);
    atomicAdd(o2 + 2, v2.z * al2); atomicAdd(o2 + 3, v2.w * al2);
}

// ---------- finalize: relu(agg + bias) ----------
__global__ void k_finalize_bf16(const float* __restrict__ agg, const float* __restrict__ bias,
                                __bf16* __restrict__ out, long long n256)
{
    long long t = (long long)blockIdx.x * blockDim.x + threadIdx.x;
    if (t >= n256) return;
    int c = (int)(t & (FEAT - 1));
    float v = agg[t] + bias[c];
    out[t] = (__bf16)(v > 0.f ? v : 0.f);
}
__global__ void k_finalize_f32(float* __restrict__ agg, const float* __restrict__ bias, long long n256)
{
    long long t = (long long)blockIdx.x * blockDim.x + threadIdx.x;
    if (t >= n256) return;
    int c = (int)(t & (FEAT - 1));
    float v = agg[t] + bias[c];
    agg[t] = v > 0.f ? v : 0.f;
}

// ---------- final linear head: wave32 per node ----------
__global__ __launch_bounds__(256) void k_fc(const float* __restrict__ h,
                                            const float* __restrict__ Wfc,
                                            const float* __restrict__ bfc,
                                            float* __restrict__ out, int n)
{
    __shared__ float Wl[FEAT * OUTD];
    __shared__ float bl[OUTD];
    for (int i = threadIdx.x; i < FEAT * OUTD; i += 256) Wl[i] = Wfc[i];
    if (threadIdx.x < OUTD) bl[threadIdx.x] = bfc[threadIdx.x];
    __syncthreads();

    int wave = threadIdx.x >> 5, lane = threadIdx.x & 31;
    int node = blockIdx.x * 8 + wave;
    if (node >= n) return;
    const float4* hp = reinterpret_cast<const float4*>(h + (size_t)node * FEAT + lane * 8);
    float4 v0 = hp[0], v1 = hp[1];
    int c = lane * 8;
#pragma unroll
    for (int j = 0; j < OUTD; ++j) {
        float p = v0.x * Wl[(c + 0) * OUTD + j] + v0.y * Wl[(c + 1) * OUTD + j]
                + v0.z * Wl[(c + 2) * OUTD + j] + v0.w * Wl[(c + 3) * OUTD + j]
                + v1.x * Wl[(c + 4) * OUTD + j] + v1.y * Wl[(c + 5) * OUTD + j]
                + v1.z * Wl[(c + 6) * OUTD + j] + v1.w * Wl[(c + 7) * OUTD + j];
#pragma unroll
        for (int off = 16; off > 0; off >>= 1) p += __shfl_down(p, off, 32);
        if (lane == 0) out[(size_t)node * OUTD + j] = p + bl[j];
    }
}

// =================== host-side orchestration ===================
extern "C" void kernel_launch(void* const* d_in, const int* in_sizes, int n_in,
                              void* d_out, int out_size, void* d_ws, size_t ws_size,
                              hipStream_t stream)
{
    const float* x   = (const float*)d_in[0];
    const int*   src = (const int*)  d_in[1];
    const int*   dst = (const int*)  d_in[2];
    const float* W1  = (const float*)d_in[3];
    const float* aS1 = (const float*)d_in[4];
    const float* aD1 = (const float*)d_in[5];
    const float* b1  = (const float*)d_in[6];
    const float* W2  = (const float*)d_in[7];
    const float* aS2 = (const float*)d_in[8];
    const float* aD2 = (const float*)d_in[9];
    const float* b2  = (const float*)d_in[10];
    const float* Wfc = (const float*)d_in[11];
    const float* bfc = (const float*)d_in[12];

    const int n = in_sizes[0] / INDIM;   // 50000
    const int e = in_sizes[1];           // 800000
    const int etot = e + n;

    // workspace carve-out
    char* w = (char*)d_ws;
    float*    hlin = (float*)w;    w += (size_t)n * FEAT * sizeof(float);
    float*    agg  = (float*)w;    w += (size_t)n * FEAT * sizeof(float);
    __bf16*   Xb   = (__bf16*)w;   w += (size_t)n * FEAT * sizeof(__bf16);
    __bf16*   Wt   = (__bf16*)w;   w += (size_t)FEAT * FEAT * sizeof(__bf16);
    float*    as_  = (float*)w;    w += (size_t)n * HEADS * sizeof(float);
    float*    ad_  = (float*)w;    w += (size_t)n * HEADS * sizeof(float);
    unsigned* smax = (unsigned*)w; w += (size_t)n * HEADS * sizeof(unsigned);
    float*    ssum = (float*)w;    w += (size_t)n * HEADS * sizeof(float);

    const dim3 blk(256);
    const dim3 gGemm(cdiv(n, TR), FEAT / TC);
    const long long n256 = (long long)n * FEAT;
    const unsigned gN256 = (unsigned)((n256 + 255) / 256);

    // ---- layer 1 ----
    k_f32_to_bf16<<<cdiv(n * INDIM, 256), blk, 0, stream>>>(x, Xb, (long long)n * INDIM);
    k_transpose_bf16<<<cdiv(INDIM * FEAT, 256), blk, 0, stream>>>(W1, Wt, INDIM);
    k_gemm_wmma<<<gGemm, blk, 0, stream>>>(Xb, Wt, hlin, n, INDIM);
    k_alpha<<<cdiv(n * HEADS, 256), blk, 0, stream>>>(hlin, aS1, aD1, as_, ad_, n);
    hipMemsetAsync(smax, 0, (size_t)n * HEADS * sizeof(unsigned), stream);
    hipMemsetAsync(ssum, 0, (size_t)n * HEADS * sizeof(float), stream);
    hipMemsetAsync(agg,  0, (size_t)n * FEAT  * sizeof(float), stream);
    k_edge_max<<<cdiv(etot * HEADS, 256), blk, 0, stream>>>(src, dst, as_, ad_, smax, e, n);
    k_edge_sum<<<cdiv(etot * HEADS, 256), blk, 0, stream>>>(src, dst, as_, ad_, smax, ssum, e, n);
    k_edge_agg<<<cdiv(etot, 8), blk, 0, stream>>>(src, dst, as_, ad_, smax, ssum, hlin, agg, e, n);
    k_finalize_bf16<<<gN256, blk, 0, stream>>>(agg, b1, Xb, n256);

    // ---- layer 2 ----
    k_transpose_bf16<<<cdiv(FEAT * FEAT, 256), blk, 0, stream>>>(W2, Wt, FEAT);
    k_gemm_wmma<<<gGemm, blk, 0, stream>>>(Xb, Wt, hlin, n, FEAT);
    k_alpha<<<cdiv(n * HEADS, 256), blk, 0, stream>>>(hlin, aS2, aD2, as_, ad_, n);
    hipMemsetAsync(smax, 0, (size_t)n * HEADS * sizeof(unsigned), stream);
    hipMemsetAsync(ssum, 0, (size_t)n * HEADS * sizeof(float), stream);
    hipMemsetAsync(agg,  0, (size_t)n * FEAT  * sizeof(float), stream);
    k_edge_max<<<cdiv(etot * HEADS, 256), blk, 0, stream>>>(src, dst, as_, ad_, smax, e, n);
    k_edge_sum<<<cdiv(etot * HEADS, 256), blk, 0, stream>>>(src, dst, as_, ad_, smax, ssum, e, n);
    k_edge_agg<<<cdiv(etot, 8), blk, 0, stream>>>(src, dst, as_, ad_, smax, ssum, hlin, agg, e, n);
    k_finalize_f32<<<gN256, blk, 0, stream>>>(agg, b2, n256);

    // ---- output head ----
    k_fc<<<cdiv(n, 8), blk, 0, stream>>>(agg, Wfc, bfc, (float*)d_out, n);
}
